// PerModalityStateTokenizer_31293131718786
// MI455X (gfx1250) — matrix operand order repaired and checked by reading, not compiled, not run on hardware
//
#include <hip/hip_runtime.h>
#include <hip/hip_bf16.h>
#include <stdint.h>

typedef _Float16 f16;
typedef __attribute__((ext_vector_type(16))) _Float16 v16h;
typedef __attribute__((ext_vector_type(8)))  _Float16 v8h;
typedef __attribute__((ext_vector_type(8)))  float    v8f;

#define M_    6
#define B_    128
#define T_    32
#define DIN_  64
#define H_    512
#define E_    512
#define NEMB_ 32

// LDS pitches in f16 elements (all 16B-multiple byte pitches)
#define XP   72     // x tile row:   64 + 8 pad  (144 B)
#define W1P  72     // W1^T row:     64 + 8 pad  (144 B)
#define HHP  520    // h row:       512 + 8 pad (1040 B)
#define W2P  40     // W2^T chunk:   32 + 8 pad  ( 80 B)

#define SMEM_HH_BYTES (32 * HHP * 2)        // 33280
#define SMEM_W1_BYTES (512 * W1P * 2)       // 73728
#define SMEM_W2_BYTES (512 * W2P * 2)       // 40960
#define SMEM_REGION   (2 * SMEM_W2_BYTES)   // 81920 >= 73728+4608
#define SMEM_TOTAL    (SMEM_HH_BYTES + SMEM_REGION)  // 115200 B

__device__ __forceinline__ v16h cat8(v8h lo, v8h hi) {
  return __builtin_shufflevector(lo, hi, 0,1,2,3,4,5,6,7,8,9,10,11,12,13,14,15);
}

// Stage one 32xE f32 chunk of W2 into LDS transposed+converted to f16:
// dst[e][k] = (f16)W2[kbase+k][e]; 4 coalesced b32 loads -> one ds_store_b64.
__device__ __forceinline__ void stage_w2(f16* __restrict__ dst,
                                         const float* __restrict__ W2g,
                                         int kc, int tid) {
  const float* src = W2g + (size_t)(kc * 32) * E_;
  #pragma unroll 4
  for (int i = 0; i < 16; ++i) {
    int q = i * 256 + tid;          // quad index 0..4095
    int e = q & (E_ - 1);
    int k = (q >> 9) * 4;           // 0,4,...,28
    float a0 = src[(size_t)(k + 0) * E_ + e];
    float a1 = src[(size_t)(k + 1) * E_ + e];
    float a2 = src[(size_t)(k + 2) * E_ + e];
    float a3 = src[(size_t)(k + 3) * E_ + e];
    union { f16 h[4]; uint64_t u; } pk;
    pk.h[0] = (f16)a0; pk.h[1] = (f16)a1; pk.h[2] = (f16)a2; pk.h[3] = (f16)a3;
    *(uint64_t*)(dst + e * W2P + k) = pk.u;
  }
}

// A operand for layer-2 chunk kc (from f16 h in LDS).
// A layout: lanes 0-15 hold K {0..15}; lanes 16-31 hold {16..31}.
__device__ __forceinline__ v16h load_a2(const f16* hh, int aRow, int hiH, int kc) {
  const f16* hp = hh + aRow * HHP + kc * 32 + hiH * 16;
  return cat8(*(const v8h*)hp, *(const v8h*)(hp + 8));
}

__global__ __launch_bounds__(256)
void tokenizer_mlp_kernel(const float* __restrict__ state,
                          const int*   __restrict__ emb_ids,
                          const float* __restrict__ W1,
                          const float* __restrict__ b1,
                          const float* __restrict__ W2,
                          const float* __restrict__ b2,
                          const float* __restrict__ type_embed,
                          float* __restrict__ out) {
  __shared__ __align__(16) unsigned char smem[SMEM_TOTAL];
  f16* hh = (f16*)smem;                               // persists both phases
  unsigned char* rgn = smem + SMEM_HH_BYTES;          // reused region
  f16* w1t  = (f16*)rgn;
  f16* xh   = (f16*)(rgn + SMEM_W1_BYTES);
  f16* w2t0 = (f16*)rgn;
  f16* w2t1 = (f16*)(rgn + SMEM_W2_BYTES);

  const int tid  = threadIdx.x;
  const int lane = tid & 31;
  const int wave = tid >> 5;
  const int l16  = lane & 15;
  const int hiH  = (lane >> 4) & 1;       // which half-wave
  const int rowBase = (wave & 1) * 16;    // 16-row half of the 32 tokens
  const int colBase = (wave >> 1) * 128;  // 128-col slab of E/H
  const int aRow = rowBase + l16;
  const int r0   = rowBase + hiH * 8;     // C/D rows this lane holds

  const int blk = blockIdx.x;
  const int b   = blk / M_;
  const int mm  = blk - b * M_;
  const int id  = emb_ids[b];

  const float* W1g = W1 + ((size_t)(mm * NEMB_ + id)) * DIN_ * H_;
  const float* W2g = W2 + ((size_t)(mm * NEMB_ + id)) * H_ * E_;
  const float* b1g = b1 + ((size_t)(mm * NEMB_ + id)) * H_;
  const float* b2g = b2 + ((size_t)(mm * NEMB_ + id)) * E_;
  const float* teg = type_embed + mm * E_;
  const float* xg  = state + ((size_t)b * T_) * (M_ * DIN_) + mm * DIN_;

  // ---- stage x (32x64) -> xh f16: float4 load, b64 store ----
  #pragma unroll
  for (int i = 0; i < 2; ++i) {
    int q = i * 256 + tid;               // quad 0..511
    int t = q >> 4;
    int d = (q & 15) * 4;
    float4 s = *(const float4*)(xg + t * (M_ * DIN_) + d);
    union { f16 h[4]; uint64_t u; } pk;
    pk.h[0] = (f16)s.x; pk.h[1] = (f16)s.y; pk.h[2] = (f16)s.z; pk.h[3] = (f16)s.w;
    *(uint64_t*)(xh + t * XP + d) = pk.u;
  }
  // ---- stage W1 (64x512) -> w1t transposed f16: w1t[e][k] ----
  #pragma unroll 4
  for (int i = 0; i < 32; ++i) {
    int q = i * 256 + tid;               // quad 0..8191
    int e = q & (H_ - 1);
    int k = (q >> 9) * 4;                // 0,4,...,60
    float a0 = W1g[(size_t)(k + 0) * H_ + e];
    float a1 = W1g[(size_t)(k + 1) * H_ + e];
    float a2 = W1g[(size_t)(k + 2) * H_ + e];
    float a3 = W1g[(size_t)(k + 3) * H_ + e];
    union { f16 h[4]; uint64_t u; } pk;
    pk.h[0] = (f16)a0; pk.h[1] = (f16)a1; pk.h[2] = (f16)a2; pk.h[3] = (f16)a3;
    *(uint64_t*)(w1t + e * W1P + k) = pk.u;
  }
  __syncthreads();

  // ---- layer 1: h = relu(x @ W1 + b1), f16 result to hh (pipelined B) ----
  v16h a1v[2];
  #pragma unroll
  for (int kb = 0; kb < 2; ++kb) {
    const f16* ap = xh + aRow * XP + kb * 32 + hiH * 16;
    a1v[kb] = cat8(*(const v8h*)ap, *(const v8h*)(ap + 8));
  }
  {
    const f16* w1c = w1t + (colBase + l16) * W1P + hiH * 16;  // tile 0 column
    v8h b00 = *(const v8h*)(w1c);
    v8h b01 = *(const v8h*)(w1c + 8);
    v8h b10 = *(const v8h*)(w1c + 32);
    v8h b11 = *(const v8h*)(w1c + 40);
    #pragma unroll
    for (int tile = 0; tile < 8; ++tile) {
      v16h bb0 = cat8(b00, b01);
      v16h bb1 = cat8(b10, b11);
      if (tile < 7) {                      // prefetch next tile's B
        const f16* nb = w1c + (tile + 1) * (16 * W1P);
        b00 = *(const v8h*)(nb);
        b01 = *(const v8h*)(nb + 8);
        b10 = *(const v8h*)(nb + 32);
        b11 = *(const v8h*)(nb + 40);
      }
      v8f c = {};
      c = __builtin_amdgcn_wmma_f32_16x16x32_f16(false, a1v[0], false, bb0,
                                                 (short)0, c, false, false);
      c = __builtin_amdgcn_wmma_f32_16x16x32_f16(false, a1v[1], false, bb1,
                                                 (short)0, c, false, false);
      int eCol = colBase + tile * 16 + l16;
      float bias = b1g[eCol];
      #pragma unroll
      for (int r = 0; r < 8; ++r) {
        float v = c[r] + bias;
        v = v > 0.0f ? v : 0.0f;
        hh[(r0 + r) * HHP + eCol] = (f16)v;
      }
    }
  }
  __syncthreads();   // hh complete; w1t/xh dead -> region reused by w2t

  // ---- layer 2: z = h @ W2, K=512 in 16 chunks, double-buffered ----
  stage_w2(w2t0, W2g, 0, tid);
  v8f acc[8];
  v8f zero = {};
  #pragma unroll
  for (int t2 = 0; t2 < 8; ++t2) acc[t2] = zero;
  __syncthreads();

  v16h av = load_a2(hh, aRow, hiH, 0);
  for (int kc = 0; kc < 16; ++kc) {
    // ---- compute first: operand loads + WMMAs issue right after barrier ----
    const f16* wb = (kc & 1) ? w2t1 : w2t0;
    const f16* wc = wb + (colBase + l16) * W2P + hiH * 16;   // tile 0 column
    v8h b0 = *(const v8h*)wc;
    v8h b1h = *(const v8h*)(wc + 8);
    v16h avNext = av;
    if (kc + 1 < 16) avNext = load_a2(hh, aRow, hiH, kc + 1);  // hh is stable
    #pragma unroll
    for (int tile = 0; tile < 8; ++tile) {
      v16h bb = cat8(b0, b1h);
      if (tile < 7) {                      // prefetch next tile's B
        const f16* nb = wc + (tile + 1) * (16 * W2P);
        b0  = *(const v8h*)nb;
        b1h = *(const v8h*)(nb + 8);
      }
      acc[tile] = __builtin_amdgcn_wmma_f32_16x16x32_f16(false, av, false, bb,
                                                         (short)0, acc[tile],
                                                         false, false);
    }
    // ---- then stage next chunk while other waves compute ----
    if (kc + 1 < 16) {
      stage_w2((kc + 1) & 1 ? w2t1 : w2t0, W2g, kc + 1, tid);
      if (kc + 2 < 16) {                   // warm L2/WGP$ one chunk further out
        const char* pf = (const char*)(W2g + (size_t)((kc + 2) * 32) * E_)
                       + (size_t)tid * 256;
        __builtin_prefetch(pf, 0, 3);
        __builtin_prefetch(pf + 128, 0, 3);
      }
    }
    __syncthreads();
    av = avNext;
  }

  // ---- epilogue: + b2 + type_embed, non-temporal write (B, M*T, E) ----
  float* og = out + ((size_t)b * (M_ * T_) + (size_t)mm * T_) * E_;
  #pragma unroll
  for (int tile = 0; tile < 8; ++tile) {
    int eCol = colBase + tile * 16 + l16;
    float add = b2g[eCol] + teg[eCol];
    #pragma unroll
    for (int r = 0; r < 8; ++r)
      __builtin_nontemporal_store(acc[tile][r] + add,
                                  &og[(size_t)(r0 + r) * E_ + eCol]);
  }
}

extern "C" void kernel_launch(void* const* d_in, const int* in_sizes, int n_in,
                              void* d_out, int out_size, void* d_ws, size_t ws_size,
                              hipStream_t stream) {
  const float* state = (const float*)d_in[0];
  const int*   ids   = (const int*)d_in[1];
  const float* W1    = (const float*)d_in[2];
  const float* b1    = (const float*)d_in[3];
  const float* W2    = (const float*)d_in[4];
  const float* b2    = (const float*)d_in[5];
  const float* te    = (const float*)d_in[6];
  float* out = (float*)d_out;

  dim3 grid(B_ * M_);
  dim3 block(256);
  hipLaunchKernelGGL(tokenizer_mlp_kernel, grid, block, 0, stream,
                     state, ids, W1, b1, W2, b2, te, out);
}
